// DecoderLayer_44032004718772
// MI455X (gfx1250) — matrix operand order repaired and checked by reading, not compiled, not run on hardware
//
#include <hip/hip_runtime.h>
#include <math.h>

// ---------------------------------------------------------------------------
// CDNA5 (gfx1250) decoder layer. All matmuls via V_WMMA_F32_16X16X32_BF16.
// wave32; 1-wave matrix workgroups (EXEC all ones around WMMA).
// 32x64 tiles per wave: B-operand reused across two row-tiles -> halves L2
// weight traffic and raises WMMA-per-load density (12 b128 loads / 8 WMMAs).
// ---------------------------------------------------------------------------

typedef __bf16 bf16_t;
typedef __attribute__((ext_vector_type(16))) __bf16 v16bf;
typedef __attribute__((ext_vector_type(8)))  __bf16 v8bf;
typedef __attribute__((ext_vector_type(8)))  float  v8f;

#define DEV __device__ __forceinline__

DEV v8f wmma_bf16(v16bf a, v16bf b, v8f c) {
  // (neg_a, A, neg_b, B, c_mod, C, reuse_a, reuse_b)
  return __builtin_amdgcn_wmma_f32_16x16x32_bf16(false, a, false, b, (short)0, c,
                                                 false, false);
}

// A operand (16x32 MxK, bf16), source row-major with K contiguous.
// ISA 7.12.2: lane L holds row M=L%16; element e -> K = 8*(L>=16) + (e<8 ? e : e+8)
DEV v16bf load_a(const bf16_t* rowptr, int kb, int half) {
  v16bf a;
  ((v8bf*)&a)[0] = *(const v8bf*)(rowptr + kb + half * 8);
  ((v8bf*)&a)[1] = *(const v8bf*)(rowptr + kb + half * 8 + 16);
  return a;
}
// B operand (32x16 KxN, bf16) fed from B^T rows (N x K row-major).
// lane L holds col N=L%16; element e -> K = 8*(L>=16) + (e<8 ? e : e+8)
DEV v16bf load_b(const bf16_t* rowptr, int kb, int half) {
  v16bf b;
  ((v8bf*)&b)[0] = *(const v8bf*)(rowptr + kb + half * 8);
  ((v8bf*)&b)[1] = *(const v8bf*)(rowptr + kb + half * 8 + 16);
  return b;
}

// ---------------------------------------------------------------------------
// fp32 -> bf16 elementwise
__global__ __launch_bounds__(256) void f2bf_kernel(const float* __restrict__ in,
                                                   bf16_t* __restrict__ out, int n) {
  int i = blockIdx.x * 256 + threadIdx.x;
  if (i < n) out[i] = (bf16_t)in[i];
}

// W (K x N fp32, row-major) -> Wt (N x K bf16, row-major)
__global__ __launch_bounds__(256) void trans_bf16_kernel(const float* __restrict__ W,
                                                         bf16_t* __restrict__ Wt,
                                                         int K, int N) {
  int idx = blockIdx.x * 256 + threadIdx.x;  // idx = n*K + k
  int n = idx / K, k = idx % K;
  Wt[idx] = (bf16_t)W[(size_t)k * N + n];
}

// ---------------------------------------------------------------------------
// GEMM: C[M x N] = A[M x K](bf16) @ B[K x N] + bias, B given as Bt (N x K bf16).
// One wave per 32x64 tile. Epilogue modes:
//   0: fp32 out (+bf16 copy if outB)        1: same with ReLU
//   2: QK head permute  col=(d*16+h) -> outB[((b*16+h)*1024+i)*64+d]
//   3: V  head permute^T col=(o*16+h) -> outB[((b*16+h)*64+o)*1024+i]
__global__ __launch_bounds__(32) void gemm_kernel(
    const bf16_t* __restrict__ A, const bf16_t* __restrict__ Bt,
    const float* __restrict__ bias, float* __restrict__ outF,
    bf16_t* __restrict__ outB, int M, int N, int K, int mode) {
  int lane = threadIdx.x;
  int half = lane >> 4, l16 = lane & 15;
  int m0 = blockIdx.x * 32, n0 = blockIdx.y * 64;

  v8f acc[2][4] = {};
  const bf16_t* arow0 = A + (size_t)(m0 + l16) * K;
  const bf16_t* arow1 = A + (size_t)(m0 + 16 + l16) * K;
  for (int k = 0; k < K; k += 32) {
    v16bf a0 = load_a(arow0, k, half);
    v16bf a1 = load_a(arow1, k, half);
#pragma unroll
    for (int t = 0; t < 4; t++) {
      const bf16_t* brow = Bt + (size_t)(n0 + t * 16 + l16) * K;
      v16bf bb = load_b(brow, k, half);
      acc[0][t] = wmma_bf16(a0, bb, acc[0][t]);
      acc[1][t] = wmma_bf16(a1, bb, acc[1][t]);
    }
  }
#pragma unroll
  for (int mt = 0; mt < 2; mt++) {
#pragma unroll
    for (int t = 0; t < 4; t++) {
#pragma unroll
      for (int r = 0; r < 8; r++) {
        int m = m0 + mt * 16 + half * 8 + r;
        int col = n0 + t * 16 + l16;
        float v = acc[mt][t][r] + bias[col];
        if (mode == 1) v = fmaxf(v, 0.0f);
        if (mode <= 1) {
          if (outF) outF[(size_t)m * N + col] = v;
          if (outB) outB[(size_t)m * N + col] = (bf16_t)v;
        } else {
          int b = m >> 10, i = m & 1023;        // rows are b*1024 + i
          int hh = col & 15, dd = col >> 4;     // head-axis-last reshape
          size_t dst = (mode == 2)
              ? (((size_t)b * 16 + hh) * 1024 + i) * 64 + dd
              : (((size_t)b * 16 + hh) * 64 + dd) * 1024 + i;
          outB[dst] = (bf16_t)v;
        }
      }
    }
  }
}

// ---------------------------------------------------------------------------
// Attention for one (b, h, 32-query block). Two-pass streaming softmax.
// Qp/Kp: [b][h][row][d=64] bf16.  Vt: [b][h][o=64][j] bf16.
// out[b][i][(o*16+h)] fp32.  Reference semantics: p = softmax(S/8) + mask; O=p@V.
__global__ __launch_bounds__(32) void attn_kernel(
    const bf16_t* __restrict__ Qp, const bf16_t* __restrict__ Kp,
    const bf16_t* __restrict__ Vt, const float* __restrict__ mask1,
    const float* __restrict__ mask2, float* __restrict__ out,
    int P_, int Cj, int E_) {
  int lane = threadIdx.x;
  int half = lane >> 4, l16 = lane & 15;
  int b = blockIdx.z, h = blockIdx.y, q0 = blockIdx.x * 32;

  const bf16_t* Q = Qp + (((size_t)b * 16 + h) * P_ + q0) * 64;
  const bf16_t* K = Kp + (((size_t)b * 16 + h) * Cj) * 64;
  const bf16_t* V = Vt + (((size_t)b * 16 + h) * 64) * (size_t)Cj;

  v16bf aq[2][2];
#pragma unroll
  for (int mt = 0; mt < 2; mt++) {
    const bf16_t* qrow = Q + (size_t)(mt * 16 + l16) * 64;
    aq[mt][0] = load_a(qrow, 0, half);
    aq[mt][1] = load_a(qrow, 32, half);
  }

  float rm[2][8], rs[2][8];
#pragma unroll
  for (int mt = 0; mt < 2; mt++)
#pragma unroll
    for (int r = 0; r < 8; r++) { rm[mt][r] = -3.0e38f; rs[mt][r] = 0.0f; }

  // ---- pass 1: online row max / sum of exp ----
  for (int j0 = 0; j0 < Cj; j0 += 16) {
    const bf16_t* krow = K + (size_t)(j0 + l16) * 64;
    v16bf bk0 = load_b(krow, 0, half);
    v16bf bk1 = load_b(krow, 32, half);
#pragma unroll
    for (int mt = 0; mt < 2; mt++) {
      v8f s = {};
      s = wmma_bf16(aq[mt][0], bk0, s);
      s = wmma_bf16(aq[mt][1], bk1, s);
      float sc[8], tm[8];
#pragma unroll
      for (int r = 0; r < 8; r++) { sc[r] = s[r] * 0.125f; tm[r] = sc[r]; }
      for (int d = 1; d < 16; d <<= 1)
#pragma unroll
        for (int r = 0; r < 8; r++) tm[r] = fmaxf(tm[r], __shfl_xor(tm[r], d, 32));
      float te[8];
#pragma unroll
      for (int r = 0; r < 8; r++) {
        float nm = fmaxf(rm[mt][r], tm[r]);
        te[r] = __expf(sc[r] - nm);
        rs[mt][r] = rs[mt][r] * __expf(rm[mt][r] - nm);
        rm[mt][r] = nm;
      }
      for (int d = 1; d < 16; d <<= 1)
#pragma unroll
        for (int r = 0; r < 8; r++) te[r] += __shfl_xor(te[r], d, 32);
#pragma unroll
      for (int r = 0; r < 8; r++) rs[mt][r] += te[r];
    }
  }
  float inv[2][8];
#pragma unroll
  for (int mt = 0; mt < 2; mt++)
#pragma unroll
    for (int r = 0; r < 8; r++) inv[mt][r] = 1.0f / rs[mt][r];

  // ---- pass 2: p = exp(s-m)/sum + mask ; O += p @ V ----
  __shared__ bf16_t pl[32][40];  // 80B row stride: 16B aligned sublanes
  v8f accO[2][4] = {};
  for (int j0 = 0; j0 < Cj; j0 += 32) {
#pragma unroll
    for (int sub = 0; sub < 2; sub++) {
      int jt = j0 + sub * 16;
      const bf16_t* krow = K + (size_t)(jt + l16) * 64;
      v16bf bk0 = load_b(krow, 0, half);
      v16bf bk1 = load_b(krow, 32, half);
      int jcol = jt + l16;
#pragma unroll
      for (int mt = 0; mt < 2; mt++) {
        v8f s = {};
        s = wmma_bf16(aq[mt][0], bk0, s);
        s = wmma_bf16(aq[mt][1], bk1, s);
#pragma unroll
        for (int r = 0; r < 8; r++) {
          int irow = q0 + mt * 16 + half * 8 + r;
          float p = __expf(s[r] * 0.125f - rm[mt][r]) * inv[mt][r];
          size_t moff = ((size_t)b * P_ + irow) * Cj + jcol;
          p += mask1[moff];
          if (mask2) p += mask2[moff];
          pl[mt * 16 + half * 8 + r][sub * 16 + l16] = (bf16_t)p;
        }
      }
    }
    __syncthreads();
    v16bf ap[2];
#pragma unroll
    for (int mt = 0; mt < 2; mt++) {
      ((v8bf*)&ap[mt])[0] = *(const v8bf*)&pl[mt * 16 + l16][half * 8];
      ((v8bf*)&ap[mt])[1] = *(const v8bf*)&pl[mt * 16 + l16][half * 8 + 16];
    }
#pragma unroll
    for (int t = 0; t < 4; t++) {
      const bf16_t* vrow = V + (size_t)(t * 16 + l16) * Cj + j0;
      v16bf bv = load_b(vrow, 0, half);
      accO[0][t] = wmma_bf16(ap[0], bv, accO[0][t]);
      accO[1][t] = wmma_bf16(ap[1], bv, accO[1][t]);
    }
    __syncthreads();
  }
#pragma unroll
  for (int mt = 0; mt < 2; mt++)
#pragma unroll
    for (int t = 0; t < 4; t++)
#pragma unroll
      for (int r = 0; r < 8; r++) {
        int irow = q0 + mt * 16 + half * 8 + r;
        int col = (t * 16 + l16) * 16 + h;  // out col = o*NH + h
        out[((size_t)b * P_ + irow) * E_ + col] = accO[mt][t][r];
      }
}

// ---------------------------------------------------------------------------
// LayerNorm(in)*g + b + res ; fp32 out (+ optional bf16 copy). One block/row.
__global__ __launch_bounds__(256) void ln_res_kernel(
    const float* __restrict__ in, const float* __restrict__ res,
    const float* __restrict__ g, const float* __restrict__ bta,
    float* __restrict__ outF, bf16_t* __restrict__ outB, int N) {
  int row = blockIdx.x, t = threadIdx.x;
  const float* rp = in + (size_t)row * N;
  float v[4];
#pragma unroll
  for (int e = 0; e < 4; e++) v[e] = rp[t + e * 256];
  __shared__ float red[256];
  red[t] = v[0] + v[1] + v[2] + v[3];
  __syncthreads();
  for (int off = 128; off; off >>= 1) {
    if (t < off) red[t] += red[t + off];
    __syncthreads();
  }
  float mean = red[0] / (float)N;
  __syncthreads();
  float d2 = 0.0f;
#pragma unroll
  for (int e = 0; e < 4; e++) { float d = v[e] - mean; d2 += d * d; }
  red[t] = d2;
  __syncthreads();
  for (int off = 128; off; off >>= 1) {
    if (t < off) red[t] += red[t + off];
    __syncthreads();
  }
  float rinv = rsqrtf(red[0] / (float)N + 1e-5f);
  const float* rr = res + (size_t)row * N;
#pragma unroll
  for (int e = 0; e < 4; e++) {
    int c = t + e * 256;
    float o = (v[e] - mean) * rinv * g[c] + bta[c] + rr[c];
    if (outF) outF[(size_t)row * N + c] = o;
    if (outB) outB[(size_t)row * N + c] = (bf16_t)o;
  }
}

// ---------------------------------------------------------------------------
extern "C" void kernel_launch(void* const* d_in, const int* in_sizes, int n_in,
                              void* d_out, int out_size, void* d_ws, size_t ws_size,
                              hipStream_t stream) {
  (void)in_sizes; (void)n_in; (void)out_size; (void)ws_size;
  const int Bb = 2, P = 1024, C = 1024, E = 1024, H = 1024, NH = 16;
  const int M = Bb * P;  // 2048 rows (also Bb*C for context)

  const float* x    = (const float*)d_in[0];
  const float* ctx  = (const float*)d_in[1];
  const float* m_causal = (const float*)d_in[2];
  const float* m_ppad   = (const float*)d_in[3];
  const float* m_cpad   = (const float*)d_in[4];
  const float* saWq = (const float*)d_in[5];  const float* sabq = (const float*)d_in[6];
  const float* saWk = (const float*)d_in[7];  const float* sabk = (const float*)d_in[8];
  const float* saWv = (const float*)d_in[9];  const float* sabv = (const float*)d_in[10];
  const float* caWq = (const float*)d_in[11]; const float* cabq = (const float*)d_in[12];
  const float* caWk = (const float*)d_in[13]; const float* cabk = (const float*)d_in[14];
  const float* caWv = (const float*)d_in[15]; const float* cabv = (const float*)d_in[16];
  const float* ln1g = (const float*)d_in[17]; const float* ln1b = (const float*)d_in[18];
  const float* ln2g = (const float*)d_in[19]; const float* ln2b = (const float*)d_in[20];
  const float* ln3g = (const float*)d_in[21]; const float* ln3b = (const float*)d_in[22];
  const float* W1 = (const float*)d_in[23];   const float* b1 = (const float*)d_in[24];
  const float* W2 = (const float*)d_in[25];   const float* b2 = (const float*)d_in[26];

  char* wp = (char*)d_ws;
  auto alloc = [&](size_t bytes) -> void* {
    void* p = (void*)wp; wp += (bytes + 255) & ~(size_t)255; return p;
  };
  bf16_t* wt[8];
  const float* wsrc[8] = {saWq, saWk, saWv, caWq, caWk, caWv, W1, W2};
  for (int i = 0; i < 8; i++) wt[i] = (bf16_t*)alloc((size_t)E * H * 2);
  bf16_t* xbf  = (bf16_t*)alloc((size_t)M * E * 2);
  bf16_t* cbf  = (bf16_t*)alloc((size_t)M * E * 2);
  bf16_t* qp   = (bf16_t*)alloc((size_t)M * H * 2);
  bf16_t* kp   = (bf16_t*)alloc((size_t)M * H * 2);
  bf16_t* vt   = (bf16_t*)alloc((size_t)M * E * 2);
  float*  tmpf = (float*)alloc((size_t)M * E * 4);  // attn out / mlp out
  float*  x1   = (float*)alloc((size_t)M * E * 4);
  bf16_t* x1bf = (bf16_t*)alloc((size_t)M * E * 2);
  float*  x2   = (float*)alloc((size_t)M * E * 4);
  bf16_t* x2bf = (bf16_t*)alloc((size_t)M * E * 2);
  bf16_t* hbf  = (bf16_t*)alloc((size_t)M * H * 2);

  dim3 b256(256), b32(32);
  for (int i = 0; i < 8; i++)
    trans_bf16_kernel<<<(E * H) / 256, b256, 0, stream>>>(wsrc[i], wt[i], E, H);
  f2bf_kernel<<<(M * E) / 256, b256, 0, stream>>>(x, xbf, M * E);
  f2bf_kernel<<<(M * E) / 256, b256, 0, stream>>>(ctx, cbf, M * E);

  dim3 ggemm(M / 32, H / 64);
  dim3 gattn(P / 32, NH, Bb);

  // ---- self attention ----
  gemm_kernel<<<ggemm, b32, 0, stream>>>(xbf, wt[0], sabq, nullptr, qp, M, H, E, 2);
  gemm_kernel<<<ggemm, b32, 0, stream>>>(xbf, wt[1], sabk, nullptr, kp, M, H, E, 2);
  gemm_kernel<<<ggemm, b32, 0, stream>>>(xbf, wt[2], sabv, nullptr, vt, M, E, E, 3);
  attn_kernel<<<gattn, b32, 0, stream>>>(qp, kp, vt, m_causal, m_ppad, tmpf, P, P, E);
  ln_res_kernel<<<M, b256, 0, stream>>>(tmpf, x, ln1g, ln1b, x1, x1bf, E);

  // ---- cross attention ----
  gemm_kernel<<<ggemm, b32, 0, stream>>>(x1bf, wt[3], cabq, nullptr, qp, M, H, E, 2);
  gemm_kernel<<<ggemm, b32, 0, stream>>>(cbf,  wt[4], cabk, nullptr, kp, M, H, E, 2);
  gemm_kernel<<<ggemm, b32, 0, stream>>>(cbf,  wt[5], cabv, nullptr, vt, M, E, E, 3);
  attn_kernel<<<gattn, b32, 0, stream>>>(qp, kp, vt, m_cpad, nullptr, tmpf, P, C, E);
  ln_res_kernel<<<M, b256, 0, stream>>>(tmpf, x1, ln2g, ln2b, x2, x2bf, E);

  // ---- MLP ----
  gemm_kernel<<<ggemm, b32, 0, stream>>>(x2bf, wt[6], b1, nullptr, hbf, M, H, E, 1);
  gemm_kernel<<<ggemm, b32, 0, stream>>>(hbf,  wt[7], b2, tmpf, nullptr, M, E, H, 0);
  ln_res_kernel<<<M, b256, 0, stream>>>(tmpf, x2, ln3g, ln3b, (float*)d_out, nullptr, E);
}
